// BiLSTM_CRF_48412871360610
// MI455X (gfx1250) — compile-verified, matrix operand order checked by reading, not run on hardware
//
#include <hip/hip_runtime.h>
#include <stdint.h>

// ---------------------------------------------------------------------------
// BiLSTM-CRF for MI455X (gfx1250, wave32, WMMA).
//  - All GEMMs use v_wmma_f32_16x16x32_f16 (f16 A/B, fp32 accum).
//  - Recurrent scan: one persistent kernel, global spin barrier per phase,
//    h-tile staged to LDS via async global->LDS copy (ASYNCcnt),
//    next-step Xg prefetched into cache (global_prefetch_b8).
//  - CRF: one wave32 per sentence, shfl-based logsumexp.
// ---------------------------------------------------------------------------

#define DEV __device__ __forceinline__

typedef _Float16 h8   __attribute__((ext_vector_type(8)));
typedef _Float16 v16h __attribute__((ext_vector_type(16)));
typedef float    v8f  __attribute__((ext_vector_type(8)));

constexpr int Bsz = 64, Tn = 256, En = 256, Hn = 512, Kt = 32;
constexpr int Gn = 4 * Hn;              // 2048 gate columns
constexpr int START_TAG = 30, END_TAG = 31;

// ---- workspace layout (bytes) ---------------------------------------------
constexpr size_t alup(size_t x) { return (x + 255) & ~(size_t)255; }
constexpr size_t OFF_XG    = 0;                                             // 2*T*B*G fp32
constexpr size_t SZ_XG     = (size_t)2 * Tn * Bsz * Gn * 4;
constexpr size_t OFF_HS    = alup(OFF_XG + SZ_XG);                          // 2*T*B*H f16
constexpr size_t SZ_HS     = (size_t)2 * Tn * Bsz * Hn * 2;
constexpr size_t OFF_XH    = alup(OFF_HS + SZ_HS);                          // B*T*E f16
constexpr size_t SZ_XH     = (size_t)Bsz * Tn * En * 2;
constexpr size_t OFF_WIH   = alup(OFF_XH + SZ_XH);                          // 2*G*E f16
constexpr size_t SZ_WIH    = (size_t)2 * Gn * En * 2;
constexpr size_t OFF_WHH   = alup(OFF_WIH + SZ_WIH);                        // 2*G*H f16
constexpr size_t SZ_WHH    = (size_t)2 * Gn * Hn * 2;
constexpr size_t OFF_WOUTT = alup(OFF_WHH + SZ_WHH);                        // K x 2H f16 (transposed)
constexpr size_t SZ_WOUTT  = (size_t)Kt * 2 * Hn * 2;
constexpr size_t OFF_BIAS  = alup(OFF_WOUTT + SZ_WOUTT);                    // 2*G fp32
constexpr size_t SZ_BIAS   = (size_t)2 * Gn * 4;
constexpr size_t OFF_HCUR  = alup(OFF_BIAS + SZ_BIAS);                      // 2*B*H f16
constexpr size_t SZ_HCUR   = (size_t)2 * Bsz * Hn * 2;
constexpr size_t OFF_CST   = alup(OFF_HCUR + SZ_HCUR);                      // 2*B*H fp32
constexpr size_t SZ_CST    = (size_t)2 * Bsz * Hn * 4;
constexpr size_t OFF_GATES = alup(OFF_CST + SZ_CST);                        // 2*B*G fp32
constexpr size_t SZ_GATES  = (size_t)2 * Bsz * Gn * 4;
constexpr size_t OFF_EMIS  = alup(OFF_GATES + SZ_GATES);                    // T*B*K fp32
constexpr size_t SZ_EMIS   = (size_t)Tn * Bsz * Kt * 4;
constexpr size_t OFF_BAR   = alup(OFF_EMIS + SZ_EMIS);                      // barrier counter

// ---- gfx1250 async global->LDS support (probe via __has_builtin) -----------
#if defined(__has_builtin)
#if __has_builtin(__builtin_amdgcn_global_load_async_to_lds_b128)
#define HAVE_ASYNC_LDS 1
#endif
#endif

// builtin parameter types (from hipcc diagnostic): param1 = AS1 int __vector(4)*,
// param2 = LDS destination; use exactly-matching GCC-style vector typedefs.
typedef int v4i_g __attribute__((vector_size(16)));
typedef __attribute__((address_space(1))) v4i_g* as1_v4i;
typedef __attribute__((address_space(3))) v4i_g* as3_v4i;

DEV void wait_asynccnt0() {
#if defined(__has_builtin) && __has_builtin(__builtin_amdgcn_s_wait_asynccnt)
  __builtin_amdgcn_s_wait_asynccnt(0);
#else
  asm volatile("s_wait_asynccnt 0" ::: "memory");
#endif
}

// ---- WMMA helpers ----------------------------------------------------------
// A frag (16x32 f16, row-major src, ld elems): lane l -> row m0+(l&15),
//   halves at k0+(l>>4)*8 and k0+(l>>4)*8+16 (per ISA 7.12.2).
// B frag (32x16): lane l -> column n0+(l&15); source must be N-major/K-contig
//   (i.e. W stored as [N][K] row-major), same per-lane chunk pattern.
DEV v16h load_frag16(const _Float16* p) {
  h8 lo = *(const h8*)(p);
  h8 hi = *(const h8*)(p + 16);
  return __builtin_shufflevector(lo, hi, 0,1,2,3,4,5,6,7,8,9,10,11,12,13,14,15);
}
DEV v8f wmma16(v16h a, v16h b, v8f c) {
  return __builtin_amdgcn_wmma_f32_16x16x32_f16(false, a, false, b, (short)0, c,
                                                false, false);
}
DEV float sigmoidf_(float x) { return 1.f / (1.f + __expf(-x)); }

// ---- persistent-kernel grid barrier (monotonic counter, re-zeroed per call)
DEV void grid_sync(unsigned* cnt, unsigned target) {
  __threadfence();
  __syncthreads();
  if (threadIdx.x == 0) {
    __hip_atomic_fetch_add(cnt, 1u, __ATOMIC_RELEASE, __HIP_MEMORY_SCOPE_AGENT);
    while (__hip_atomic_load(cnt, __ATOMIC_ACQUIRE, __HIP_MEMORY_SCOPE_AGENT) <
           target) {
      __builtin_amdgcn_s_sleep(1);
    }
  }
  __syncthreads();
  __threadfence();
}

// ---------------------------------------------------------------------------
// Kernel 1: convert inputs to f16, fuse biases, transpose W_out, zero state.
// ---------------------------------------------------------------------------
__global__ void prep_kernel(const float* __restrict__ X,
                            const float* __restrict__ Wih_f, const float* __restrict__ Whh_f,
                            const float* __restrict__ bih_f, const float* __restrict__ bhh_f,
                            const float* __restrict__ Wih_b, const float* __restrict__ Whh_b,
                            const float* __restrict__ bih_b, const float* __restrict__ bhh_b,
                            const float* __restrict__ Wout,
                            _Float16* __restrict__ Xh, _Float16* __restrict__ Wih16,
                            _Float16* __restrict__ Whh16, _Float16* __restrict__ WoutT,
                            float* __restrict__ bias, _Float16* __restrict__ h_cur,
                            float* __restrict__ c_state, unsigned* __restrict__ bar) {
  size_t tid = (size_t)blockIdx.x * blockDim.x + threadIdx.x;
  size_t nth = (size_t)gridDim.x * blockDim.x;
  for (size_t i = tid; i < (size_t)Bsz * Tn * En; i += nth) Xh[i] = (_Float16)X[i];
  for (size_t i = tid; i < (size_t)Gn * En; i += nth) {
    Wih16[i] = (_Float16)Wih_f[i];
    Wih16[(size_t)Gn * En + i] = (_Float16)Wih_b[i];
  }
  for (size_t i = tid; i < (size_t)Gn * Hn; i += nth) {
    Whh16[i] = (_Float16)Whh_f[i];
    Whh16[(size_t)Gn * Hn + i] = (_Float16)Whh_b[i];
  }
  for (size_t i = tid; i < (size_t)Kt * 2 * Hn; i += nth) {
    size_t n = i / (2 * Hn), k = i % (2 * Hn);
    WoutT[i] = (_Float16)Wout[k * Kt + n];       // [K][2H] <- [2H][K]
  }
  for (size_t i = tid; i < (size_t)Gn; i += nth) {
    bias[i]      = bih_f[i] + bhh_f[i];
    bias[Gn + i] = bih_b[i] + bhh_b[i];
  }
  for (size_t i = tid; i < (size_t)2 * Bsz * Hn; i += nth) {
    h_cur[i] = (_Float16)0.f;
    c_state[i] = 0.f;
  }
  if (tid == 0) *bar = 0u;
}

// ---------------------------------------------------------------------------
// Kernel 2: Xg[dir][t][b][g] = sum_k X[b][tsrc][k] * W_ih[dir][g][k]
//   (tsrc = t for fwd, T-1-t for bwd scan order). One wave -> 16x64 tile.
// ---------------------------------------------------------------------------
__global__ void __launch_bounds__(256) xg_gemm_kernel(const _Float16* __restrict__ Xh,
                                                      const _Float16* __restrict__ Wih16,
                                                      float* __restrict__ Xg) {
  const int wave = (int)((blockIdx.x * blockDim.x + threadIdx.x) >> 5);
  const int lane = threadIdx.x & 31;
  const int tilesPerDir = (Tn * Bsz / 16) * (Gn / 64);   // 1024 * 32
  const int dir = wave / tilesPerDir;
  if (dir >= 2) return;
  const int tw = wave % tilesPerDir;
  const int mt = tw / (Gn / 64), nt = tw % (Gn / 64);
  const int m0 = mt * 16, n0 = nt * 64;
  const int l15 = lane & 15, half = lane >> 4;

  const _Float16* Wd = Wih16 + (size_t)dir * Gn * En;
  float* Xgd = Xg + (size_t)dir * Tn * Bsz * Gn;

  const int m = m0 + l15;                // t-major row: m = t*B + b
  const int t = m >> 6, b = m & 63;
  const int tsrc = dir ? (Tn - 1 - t) : t;
  const _Float16* arow = Xh + ((size_t)b * Tn + tsrc) * En + half * 8;
  const _Float16* bp = Wd + (size_t)(n0 + l15) * En + half * 8;

  v8f a0 = {}, a1 = {}, a2 = {}, a3 = {};
#pragma unroll
  for (int k0 = 0; k0 < En; k0 += 32) {
    v16h a  = load_frag16(arow + k0);
    v16h b0 = load_frag16(bp + k0);
    v16h b1 = load_frag16(bp + k0 + (size_t)16 * En);
    v16h b2 = load_frag16(bp + k0 + (size_t)32 * En);
    v16h b3 = load_frag16(bp + k0 + (size_t)48 * En);
    a0 = wmma16(a, b0, a0);
    a1 = wmma16(a, b1, a1);
    a2 = wmma16(a, b2, a2);
    a3 = wmma16(a, b3, a3);
  }
#pragma unroll
  for (int r = 0; r < 8; ++r) {
    int row = m0 + r + half * 8;
    float* orow = Xgd + (size_t)row * Gn;
    orow[n0 + l15]      = a0[r];
    orow[n0 + 16 + l15] = a1[r];
    orow[n0 + 32 + l15] = a2[r];
    orow[n0 + 48 + l15] = a3[r];
  }
}

// ---------------------------------------------------------------------------
// Kernel 3: persistent bidirectional LSTM scan.
//   grid = 64 blocks x 256 threads; blocks [0,32) = fwd, [32,64) = bwd.
//   All 8 waves of a block share one 16-row M-tile of h: stage it in LDS
//   once per step via async global->LDS copy; B (Whh) streams from L2.
//   Per step: phase A gates = Xg[t] + bias + h@Whh^T (WMMA), grid_sync,
//             phase B pointwise LSTM cell update, grid_sync.
// ---------------------------------------------------------------------------
__global__ void __launch_bounds__(256) lstm_scan_kernel(
    const float* __restrict__ Xg, const _Float16* __restrict__ Whh16,
    const float* __restrict__ bias, _Float16* __restrict__ h_cur,
    float* __restrict__ c_state, float* __restrict__ gates,
    _Float16* __restrict__ hs, unsigned* __restrict__ bar) {
  __shared__ _Float16 sA[16 * Hn];                       // 16 KB h tile
  const int dir = blockIdx.x >> 5;
  const int blkInDir = blockIdx.x & 31;
  const int lane = threadIdx.x & 31;
  const int wave = blkInDir * 8 + (threadIdx.x >> 5);    // 0..255 per dir
  const int l15 = lane & 15, half = lane >> 4;

  const _Float16* Wd  = Whh16 + (size_t)dir * Gn * Hn;
  const float*    bd  = bias + dir * Gn;
  _Float16*       hc  = h_cur + dir * Bsz * Hn;
  float*          cs  = c_state + dir * Bsz * Hn;
  float*          gd  = gates + dir * Bsz * Gn;
  _Float16*       hsd = hs + (size_t)dir * Tn * Bsz * Hn;
  const float*    Xgd = Xg + (size_t)dir * Tn * Bsz * Gn;

  // 512 16x16 gate tiles per dir (4 M-tiles x 128 N-tiles); 2 per wave.
  // All waves of a block share mt == blkInDir>>3.
  const int tile0 = wave * 2;
  const int m0 = (tile0 >> 7) * 16;                      // == (blkInDir>>3)*16
  const int n0 = (tile0 & 127) * 16;
  const int tidInDir = blkInDir * 256 + (int)threadIdx.x;

  const _Float16* hrows = hc + (size_t)m0 * Hn;          // 16 KB contiguous
  const _Float16* b0p   = Wd + (size_t)(n0 + l15) * Hn + half * 8;
  const _Float16* b1p   = b0p + (size_t)16 * Hn;
  const _Float16* aL    = sA + (size_t)l15 * Hn + half * 8;

  for (int t = 0; t < Tn; ++t) {
    // ---- stage h tile (h_{t-1}) into LDS -------------------------------
#if defined(HAVE_ASYNC_LDS)
#pragma unroll
    for (int o = 0; o < 4; ++o) {
      int off = ((int)threadIdx.x + o * 256) * 8;        // 8 halfs = 16 B
      __builtin_amdgcn_global_load_async_to_lds_b128(
          (as1_v4i)(hrows + off), (as3_v4i)(sA + off), 0, 0);
    }
    wait_asynccnt0();
#else
#pragma unroll
    for (int o = 0; o < 4; ++o) {
      int off = ((int)threadIdx.x + o * 256) * 8;
      *(h8*)(sA + off) = *(const h8*)(hrows + off);
    }
#endif
    __syncthreads();

    const float* xgt = Xgd + (size_t)t * Bsz * Gn;
    v8f c0, c1;
#pragma unroll
    for (int r = 0; r < 8; ++r) {
      int row = m0 + r + half * 8;
      c0[r] = xgt[(size_t)row * Gn + n0 + l15]      + bd[n0 + l15];
      c1[r] = xgt[(size_t)row * Gn + n0 + 16 + l15] + bd[n0 + 16 + l15];
    }
    // prefetch next step's Xg tile into cache while we compute
    if (t + 1 < Tn) {
      const float* xg1 = xgt + (size_t)Bsz * Gn;
#pragma unroll
      for (int r = 0; r < 8; ++r) {
        int row = m0 + r + half * 8;
        __builtin_prefetch(xg1 + (size_t)row * Gn + n0 + l15, 0, 0);
      }
    }
#pragma unroll 4
    for (int k0 = 0; k0 < Hn; k0 += 32) {
      v16h a  = load_frag16(aL + k0);                    // LDS (ds_load)
      v16h b0 = load_frag16(b0p + k0);
      v16h b1 = load_frag16(b1p + k0);
      c0 = wmma16(a, b0, c0);
      c1 = wmma16(a, b1, c1);
    }
#pragma unroll
    for (int r = 0; r < 8; ++r) {
      int row = m0 + r + half * 8;
      gd[(size_t)row * Gn + n0 + l15]      = c0[r];
      gd[(size_t)row * Gn + n0 + 16 + l15] = c1[r];
    }
    grid_sync(bar, (unsigned)(2 * t + 1) * gridDim.x);

    for (int e = tidInDir; e < Bsz * Hn; e += 32 * 256) {
      int b = e >> 9, j = e & (Hn - 1);
      const float* gb = gd + (size_t)b * Gn;
      float ig = sigmoidf_(gb[j]);
      float fg = sigmoidf_(gb[Hn + j]);
      float gg = tanhf(gb[2 * Hn + j]);
      float og = sigmoidf_(gb[3 * Hn + j]);
      float c  = fg * cs[e] + ig * gg;
      cs[e] = c;
      float h = og * tanhf(c);
      hc[e] = (_Float16)h;
      hsd[(size_t)t * Bsz * Hn + e] = (_Float16)h;
    }
    grid_sync(bar, (unsigned)(2 * t + 2) * gridDim.x);
  }
}

// ---------------------------------------------------------------------------
// Kernel 4: emissions[t*B+b][k] = [hf(t) ; hb(T-1-t)] @ W_out + b_out
//   One wave -> 16x32 tile (whole K dim), K-loop over 1024.
// ---------------------------------------------------------------------------
__global__ void __launch_bounds__(256) emis_gemm_kernel(const _Float16* __restrict__ hs,
                                                        const _Float16* __restrict__ WoutT,
                                                        const float* __restrict__ bout,
                                                        float* __restrict__ emis) {
  const int wave = (int)((blockIdx.x * blockDim.x + threadIdx.x) >> 5);
  const int lane = threadIdx.x & 31;
  if (wave >= Tn * Bsz / 16) return;
  const int m0 = wave * 16;
  const int l15 = lane & 15, half = lane >> 4;
  const int m = m0 + l15, t = m >> 6, b = m & 63;
  const _Float16* hf = hs;
  const _Float16* hb = hs + (size_t)Tn * Bsz * Hn;
  const _Float16* af = hf + ((size_t)t * Bsz + b) * Hn + half * 8;
  const _Float16* ab = hb + ((size_t)(Tn - 1 - t) * Bsz + b) * Hn + half * 8;
  const _Float16* bp = WoutT + (size_t)l15 * (2 * Hn) + half * 8;

  v8f a0 = {}, a1 = {};
#pragma unroll 4
  for (int k0 = 0; k0 < 2 * Hn; k0 += 32) {
    const _Float16* arow = (k0 < Hn) ? (af + k0) : (ab + (k0 - Hn));
    v16h a  = load_frag16(arow);
    v16h b0 = load_frag16(bp + k0);
    v16h b1 = load_frag16(bp + k0 + (size_t)16 * 2 * Hn);
    a0 = wmma16(a, b0, a0);
    a1 = wmma16(a, b1, a1);
  }
#pragma unroll
  for (int r = 0; r < 8; ++r) {
    int row = m0 + r + half * 8;               // row = t*B + b
    emis[(size_t)row * Kt + l15]      = a0[r] + bout[l15];
    emis[(size_t)row * Kt + 16 + l15] = a1[r] + bout[16 + l15];
  }
}

// ---------------------------------------------------------------------------
// Kernel 5: CRF — one wave32 per sentence. Lane j tracks tag-j alpha.
// ---------------------------------------------------------------------------
__global__ void __launch_bounds__(1024) crf_kernel(const float* __restrict__ emis,
                                                   const int* __restrict__ tags,
                                                   const float* __restrict__ masks,
                                                   const float* __restrict__ trans,
                                                   float* __restrict__ out) {
  __shared__ float sT[Kt * Kt];
  for (int i = threadIdx.x; i < Kt * Kt; i += blockDim.x) sT[i] = trans[i];
  __syncthreads();
  const int wave = threadIdx.x >> 5;
  const int lane = threadIdx.x & 31;
  const int b = blockIdx.x * (blockDim.x >> 5) + wave;
  if (b >= Bsz) return;

  const int j = lane;
  float alpha = sT[START_TAG * Kt + j] + emis[(size_t)b * Kt + j];
  for (int t = 1; t < Tn; ++t) {
    float e = emis[(size_t)t * Bsz * Kt + b * Kt + j];
    float mmax = -__builtin_inff();
#pragma unroll 4
    for (int i = 0; i < Kt; ++i) {
      float v = __shfl(alpha, i, 32) + sT[i * Kt + j];
      mmax = fmaxf(mmax, v);
    }
    float s = 0.f;
#pragma unroll 4
    for (int i = 0; i < Kt; ++i) {
      float v = __shfl(alpha, i, 32) + sT[i * Kt + j];
      s += __expf(v - mmax);
    }
    float na = mmax + __logf(s) + e;
    float mk = masks[b * Tn + t];
    alpha = na * mk + alpha * (1.f - mk);
  }
  alpha += sT[j * Kt + END_TAG];
  float mx = alpha;
  for (int o = 16; o > 0; o >>= 1) mx = fmaxf(mx, __shfl_xor(mx, o, 32));
  float se = __expf(alpha - mx);
  for (int o = 16; o > 0; o >>= 1) se += __shfl_xor(se, o, 32);
  float logZ = mx + __logf(se);

  if (lane == 0) {
    int tg0 = tags[b * Tn];
    float score = sT[START_TAG * Kt + tg0] + emis[(size_t)b * Kt + tg0];
    int prev = tg0;
    float msum = masks[b * Tn];
    for (int t = 1; t < Tn; ++t) {
      int tg = tags[b * Tn + t];
      float mk = masks[b * Tn + t];
      float e = emis[(size_t)t * Bsz * Kt + b * Kt + tg];
      score += (e + sT[prev * Kt + tg]) * mk;
      prev = tg;
      msum += mk;
    }
    int last_idx = (int)msum - 1;
    score += sT[tags[b * Tn + last_idx] * Kt + END_TAG];
    out[b] = logZ - score;
  }
}

// ---------------------------------------------------------------------------
extern "C" void kernel_launch(void* const* d_in, const int* in_sizes, int n_in,
                              void* d_out, int out_size, void* d_ws, size_t ws_size,
                              hipStream_t stream) {
  (void)in_sizes; (void)n_in; (void)out_size; (void)ws_size;
  const float* X     = (const float*)d_in[0];
  const int*   tags  = (const int*)  d_in[1];
  const float* masks = (const float*)d_in[2];
  const float* Wih_f = (const float*)d_in[3];
  const float* Whh_f = (const float*)d_in[4];
  const float* bih_f = (const float*)d_in[5];
  const float* bhh_f = (const float*)d_in[6];
  const float* Wih_b = (const float*)d_in[7];
  const float* Whh_b = (const float*)d_in[8];
  const float* bih_b = (const float*)d_in[9];
  const float* bhh_b = (const float*)d_in[10];
  const float* Wout  = (const float*)d_in[11];
  const float* bout  = (const float*)d_in[12];
  const float* trans = (const float*)d_in[13];
  float* out = (float*)d_out;

  char* w = (char*)d_ws;        // requires ws_size >= ~306 MB (Xg fp32 dominates)
  float*    Xg     = (float*)(w + OFF_XG);
  _Float16* hs     = (_Float16*)(w + OFF_HS);
  _Float16* Xh     = (_Float16*)(w + OFF_XH);
  _Float16* Wih16  = (_Float16*)(w + OFF_WIH);
  _Float16* Whh16  = (_Float16*)(w + OFF_WHH);
  _Float16* WoutT  = (_Float16*)(w + OFF_WOUTT);
  float*    bias   = (float*)(w + OFF_BIAS);
  _Float16* hcur   = (_Float16*)(w + OFF_HCUR);
  float*    cst    = (float*)(w + OFF_CST);
  float*    gates  = (float*)(w + OFF_GATES);
  float*    emis   = (float*)(w + OFF_EMIS);
  unsigned* bar    = (unsigned*)(w + OFF_BAR);

  prep_kernel<<<1024, 256, 0, stream>>>(X, Wih_f, Whh_f, bih_f, bhh_f,
                                        Wih_b, Whh_b, bih_b, bhh_b, Wout,
                                        Xh, Wih16, Whh16, WoutT, bias, hcur, cst, bar);
  // 2 dirs * 1024 M-tiles * 32 N-tiles = 65536 waves -> 8192 blocks of 8 waves
  xg_gemm_kernel<<<8192, 256, 0, stream>>>(Xh, Wih16, Xg);
  // persistent scan: 64 blocks (32 per direction), 8 waves each
  lstm_scan_kernel<<<64, 256, 0, stream>>>(Xg, Whh16, bias, hcur, cst, gates, hs, bar);
  // 1024 row-tiles, 1 wave each -> 128 blocks
  emis_gemm_kernel<<<128, 256, 0, stream>>>(hs, WoutT, bout, emis);
  crf_kernel<<<2, 1024, 0, stream>>>(emis, tags, masks, trans, out);
}
